// TransRelation_412316861005
// MI455X (gfx1250) — matrix operand order repaired and checked by reading, not compiled, not run on hardware
//
#include <hip/hip_runtime.h>
#include <hip/hip_bf16.h>
#include <math.h>
#include <stdint.h>

typedef __attribute__((ext_vector_type(16))) _Float16 v16h;
typedef __attribute__((ext_vector_type(8)))  _Float16 v8h;
typedef __attribute__((ext_vector_type(4)))  _Float16 v4h;
typedef __attribute__((ext_vector_type(8)))  float    v8f;
typedef __attribute__((ext_vector_type(4)))  float    v4f;

#define B_    4
#define T_    1024
#define S_    2048
#define HT    2048
#define DR    512
#define NKV_  8
#define HD_   64
#define NQ_   32
#define GQ    4
#define SCALE_ 0.125f      /* 1/sqrt(64) */
#define ALPHA_ 2.0f
#define LN_EPS_ 1e-5f

#define ASZ32 (16 * 32 * 16)   /* A fragment-major sub-tile (K-chunk of 32), halves */
#define BSZ32 (4 * 32 * 16)    /* B fragment-major sub-tile, halves */
#define ATS   (2 * ASZ32)      /* A tile for K-step 64 */
#define BTS   (2 * BSZ32)      /* B tile for K-step 64 */

// ---------------------------------------------------------------------------
// CDNA5 async LDS copy (ASYNCcnt-tracked, bypasses VGPRs)
// ---------------------------------------------------------------------------
__device__ __forceinline__ void async_b128(uint32_t lds, const void* g)
{
    asm volatile("global_load_async_to_lds_b128 %0, %1, off"
                 :: "v"(lds), "v"(g) : "memory");
}
__device__ __forceinline__ void wait_async0()
{
#if __has_builtin(__builtin_amdgcn_s_wait_asynccnt)
    __builtin_amdgcn_s_wait_asynccnt(0);
#else
    asm volatile("s_wait_asynccnt 0x0" ::: "memory");
#endif
}

// ---------------------------------------------------------------------------
// stage one 256x64 A tile + 64x64 B tile into swizzled fragment-major LDS,
// entirely with per-lane async b128 copies (swizzle folded into addressing).
// Ah: f16 [M][K] row-major.  Wt: f16 [N][K] (pre-transposed weights).
// ---------------------------------------------------------------------------
__device__ __forceinline__ void gemm_stage_async(const _Float16* __restrict__ Ah,
                                                 const _Float16* __restrict__ Wt,
                                                 _Float16* Asb, _Float16* Bsb,
                                                 int m0, int n0, int k0, int K, int tid)
{
    #pragma unroll
    for (int it = 0; it < 8; ++it) {          // A: 2048 chunks of 16B
        int cid = it * 256 + tid;
        int r = cid >> 3, q = cid & 7;
        int kj = q >> 2, hi = q & 1, o = (q >> 1) & 1;
        const _Float16* src = Ah + (size_t)(m0 + r) * K + k0 + 32 * kj + 8 * hi + 16 * o;
        uint32_t dst = (uint32_t)(uintptr_t)
            &Asb[kj * ASZ32 + (((r >> 4) * 32) + (r & 15) + 16 * hi) * 16 + 8 * o];
        async_b128(dst, src);
    }
    #pragma unroll
    for (int it = 0; it < 2; ++it) {          // B: 512 chunks of 16B
        int cid = it * 256 + tid;
        int c = cid >> 3, q = cid & 7;
        int kj = q >> 2, hi = q & 1, o = (q >> 1) & 1;
        const _Float16* src = Wt + (size_t)(n0 + c) * K + k0 + 32 * kj + 8 * hi + 16 * o;
        uint32_t dst = (uint32_t)(uintptr_t)
            &Bsb[kj * BSZ32 + (((c >> 4) * 32) + (c & 15) + 16 * hi) * 16 + 8 * o];
        async_b128(dst, src);
    }
}

// ---------------------------------------------------------------------------
// All-f16 GEMM: C[M,N] = Ah[M,K] @ Wt[N,K]^T + bias.
// Double-buffered async-staged LDS; fragments are contiguous 32B LDS reads.
// Output modes: outH (f16 row-major) / outT (f16 transposed, for V) /
//               outF = alpha*resid + val (f32, for O-proj).
// Block: 256 thr (8 waves). Block tile 256(M) x 64(N); K-step 64 (16 WMMA).
// ---------------------------------------------------------------------------
__global__ __launch_bounds__(256)
void gemm_wmma_async_kernel(const _Float16* __restrict__ Ah,
                            const _Float16* __restrict__ Wt,
                            const float* __restrict__ bias,
                            _Float16* __restrict__ outH, float* __restrict__ outF,
                            const float* __restrict__ resid, float alpha,
                            _Float16* __restrict__ outT, int Srows,
                            int M, int N, int K)
{
    __shared__ __align__(32) _Float16 As[2 * ATS];   // 64 KB
    __shared__ __align__(32) _Float16 Bs[2 * BTS];   // 16 KB

    const int tid  = threadIdx.x;
    const int lane = tid & 31;
    const int wave = tid >> 5;
    const int m0   = blockIdx.y * 256;
    const int n0   = blockIdx.x * 64;

    v8f acc[2][4];
    #pragma unroll
    for (int mi = 0; mi < 2; ++mi)
        #pragma unroll
        for (int nt = 0; nt < 4; ++nt)
            #pragma unroll
            for (int e = 0; e < 8; ++e) acc[mi][nt][e] = 0.0f;

    // prologue: stage tile 0 into buffer 0
    gemm_stage_async(Ah, Wt, As, Bs, m0, n0, 0, K, tid);
    wait_async0();
    __syncthreads();

    int buf = 0;
    for (int k0 = 0; k0 < K; k0 += 64) {
        const bool has_next = (k0 + 64) < K;
        if (has_next)
            gemm_stage_async(Ah, Wt, As + (buf ^ 1) * ATS, Bs + (buf ^ 1) * BTS,
                             m0, n0, k0 + 64, K, tid);

        const _Float16* Ac = As + buf * ATS;
        const _Float16* Bc = Bs + buf * BTS;
        v16h af[2][2];   // [kj][mi]
        #pragma unroll
        for (int kj = 0; kj < 2; ++kj)
            #pragma unroll
            for (int mi = 0; mi < 2; ++mi)
                af[kj][mi] = *(const v16h*)
                    &Ac[kj * ASZ32 + ((wave * 2 + mi) * 32 + lane) * 16];
        #pragma unroll
        for (int nt = 0; nt < 4; ++nt) {
            #pragma unroll
            for (int kj = 0; kj < 2; ++kj) {
                v16h bf = *(const v16h*)&Bc[kj * BSZ32 + (nt * 32 + lane) * 16];
                acc[0][nt] = __builtin_amdgcn_wmma_f32_16x16x32_f16(
                    false, af[kj][0], false, bf, (short)0, acc[0][nt], false, false);
                acc[1][nt] = __builtin_amdgcn_wmma_f32_16x16x32_f16(
                    false, af[kj][1], false, bf, (short)0, acc[1][nt], false, false);
            }
        }

        wait_async0();       // next-tile copies landed (overlapped with WMMAs)
        __syncthreads();
        buf ^= 1;
    }

    // ---- epilogue (C layout: VGPR r -> row r (+8 hi half); lane&15 -> col)
    const int row_off = (lane >> 4) ? 8 : 0;
    #pragma unroll
    for (int mi = 0; mi < 2; ++mi) {
        #pragma unroll
        for (int nt = 0; nt < 4; ++nt) {
            #pragma unroll
            for (int r = 0; r < 8; ++r) {
                int grow = m0 + (wave * 2 + mi) * 16 + row_off + r;
                int gcol = n0 + nt * 16 + (lane & 15);
                float val = acc[mi][nt][r] + bias[gcol];
                if (outH) {
                    outH[(size_t)grow * N + gcol] = (_Float16)val;
                } else if (outT) {
                    int bb = grow / Srows, ss = grow - bb * Srows;
                    outT[((size_t)bb * N + gcol) * Srows + ss] = (_Float16)val;
                } else {
                    size_t o = (size_t)grow * N + gcol;
                    outF[o] = alpha * resid[o] + val;
                }
            }
        }
    }
}

// ---------------------------------------------------------------------------
// one-time f32 -> f16 convert (vectorized)
// ---------------------------------------------------------------------------
__global__ void cvt_f16_kernel(const float* __restrict__ x, _Float16* __restrict__ y,
                               int n)
{
    int i = (blockIdx.x * blockDim.x + threadIdx.x) * 8;
    if (i >= n) return;
    v4f a = *(const v4f*)(x + i);
    v4f b = *(const v4f*)(x + i + 4);
    v8h h;
    #pragma unroll
    for (int j = 0; j < 4; ++j) { h[j] = (_Float16)a[j]; h[4 + j] = (_Float16)b[j]; }
    *(v8h*)(y + i) = h;
}

// ---------------------------------------------------------------------------
// one-time weight transpose+convert: W[K][N] f32 -> Wt[N][K] f16 (LDS tiled)
// ---------------------------------------------------------------------------
__global__ __launch_bounds__(256)
void wtrans_kernel(const float* __restrict__ W, _Float16* __restrict__ Wt,
                   int K, int N)
{
    __shared__ _Float16 tile[64 * 40];   // [n][k], row stride 40 (16B-aligned)
    int k0 = blockIdx.y * 32;
    int n0 = blockIdx.x * 64;
    int tid = threadIdx.x;
    #pragma unroll
    for (int it = 0; it < 8; ++it) {
        int idx = it * 256 + tid;
        int kk = idx >> 6, nn = idx & 63;
        tile[nn * 40 + kk] = (_Float16)W[(size_t)(k0 + kk) * N + n0 + nn];
    }
    __syncthreads();
    int nn = tid >> 2, oct = tid & 3;
    *(v8h*)&Wt[(size_t)(n0 + nn) * K + k0 + oct * 8] = *(const v8h*)&tile[nn * 40 + oct * 8];
}

// ---------------------------------------------------------------------------
// RoPE (with optional even/odd KV-head interleave permutation). Vector I/O.
// ---------------------------------------------------------------------------
__global__ void rope_kernel(const _Float16* __restrict__ x, _Float16* __restrict__ y,
                            const float* __restrict__ embA, const float* __restrict__ embB,
                            int rows, int nheads, int permute)
{
    int tid = blockIdx.x * blockDim.x + threadIdx.x;
    if (tid >= rows * nheads) return;
    int row = tid / nheads;
    int ho  = tid - row * nheads;
    int hs  = ho;
    const float* emb = embA;
    if (permute) {
        if (ho < nheads / 2) { hs = 2 * ho;                    emb = embA; }
        else                 { hs = 2 * (ho - nheads / 2) + 1; emb = embB; }
    }
    const _Float16* xs = x + ((size_t)row * nheads + hs) * HD_;
    _Float16*       yd = y + ((size_t)row * nheads + ho) * HD_;
    const float*    e  = emb + (size_t)row * HD_;

    v16h xv[4];
    #pragma unroll
    for (int i = 0; i < 4; ++i) xv[i] = *(const v16h*)(xs + 16 * i);
    float xf[64];
    #pragma unroll
    for (int i = 0; i < 4; ++i)
        #pragma unroll
        for (int j = 0; j < 16; ++j) xf[16 * i + j] = (float)xv[i][j];

    float se[32], ce[32];
    #pragma unroll
    for (int i = 0; i < 8; ++i) {
        v4f s4 = *(const v4f*)(e + 4 * i);
        v4f c4 = *(const v4f*)(e + 32 + 4 * i);
        #pragma unroll
        for (int j = 0; j < 4; ++j) { se[4 * i + j] = s4[j]; ce[4 * i + j] = c4[j]; }
    }

    float o[64];
    #pragma unroll
    for (int i = 0; i < 32; ++i) {
        float x0 = xf[2 * i], x1 = xf[2 * i + 1];
        o[i]      = x0 * ce[i] - x1 * se[i];
        o[32 + i] = x1 * ce[i] + x0 * se[i];
    }
    v16h yv[4];
    #pragma unroll
    for (int i = 0; i < 4; ++i)
        #pragma unroll
        for (int j = 0; j < 16; ++j) yv[i][j] = (_Float16)o[16 * i + j];
    #pragma unroll
    for (int i = 0; i < 4; ++i) *(v16h*)(yd + 16 * i) = yv[i];
}

// ---------------------------------------------------------------------------
// Flash attention: 1 wave per 16-row Q tile per (b, kv-head, group).
// K/V fragments preloaded per chunk (b128, V pre-transposed); P goes through
// LDS in A-fragment layout.  Output written as f16 for the O-projection.
// ---------------------------------------------------------------------------
__global__ __launch_bounds__(32)
void flash_attn_kernel(const _Float16* __restrict__ q, const _Float16* __restrict__ k,
                       const _Float16* __restrict__ vt, _Float16* __restrict__ out)
{
    __shared__ float row_m[16];
    __shared__ float row_l[16];
    __shared__ __align__(32) _Float16 ps[32 * 16];

    int bid = blockIdx.x;
    int tt  = bid & 63;
    int tmp = bid >> 6;
    int g   = tmp & 3;  tmp >>= 2;
    int kvh = tmp & 7;
    int b   = tmp >> 3;
    int qh  = kvh * GQ + g;
    int t0  = tt * 16;

    const int lane = threadIdx.x;
    const int col  = lane & 15;
    const int hi   = lane >> 4;

    if (lane < 16) { row_m[lane] = -1e30f; row_l[lane] = 0.0f; }
    __syncthreads();

    v16h qa[2];
    {
        int kb = hi ? 16 : 0;
        const _Float16* qp = q + (((size_t)b * T_ + (t0 + col)) * NQ_ + qh) * HD_;
        #pragma unroll
        for (int j = 0; j < 2; ++j) {
            v8h lo = *(const v8h*)(qp + 32 * j + kb);
            v8h hh = *(const v8h*)(qp + 32 * j + kb + 8);
            v16h f = __builtin_shufflevector(lo, hh, 0, 1, 2, 3, 4, 5, 6, 7,
                                             8, 9, 10, 11, 12, 13, 14, 15);
            #pragma unroll
            for (int e = 0; e < 16; ++e)
                qa[j][e] = (_Float16)((float)f[e] * SCALE_);
        }
    }

    v8f oacc[4];
    #pragma unroll
    for (int nt = 0; nt < 4; ++nt)
        #pragma unroll
        for (int e = 0; e < 8; ++e) oacc[nt][e] = 0.0f;

    const int rbase = hi ? 8 : 0;
    const int b_kb  = hi ? 16 : 0;
    const _Float16* vbase = vt + ((size_t)b * DR + kvh * HD_) * S_;

    for (int sc = 0; sc < S_; sc += 32) {
        v16h kf[2][2];
        #pragma unroll
        for (int nt = 0; nt < 2; ++nt) {
            const _Float16* kp =
                k + (((size_t)b * S_ + (sc + nt * 16 + col)) * NKV_ + kvh) * HD_;
            kf[nt][0] = *(const v16h*)(kp + b_kb);
            kf[nt][1] = *(const v16h*)(kp + 32 + b_kb);
        }
        v16h vf[4];
        #pragma unroll
        for (int nt = 0; nt < 4; ++nt)
            vf[nt] = *(const v16h*)(vbase + (size_t)(nt * 16 + col) * S_ + sc + b_kb);

        v8f sacc[2];
        #pragma unroll
        for (int nt = 0; nt < 2; ++nt)
            #pragma unroll
            for (int e = 0; e < 8; ++e) sacc[nt][e] = 0.0f;
        #pragma unroll
        for (int nt = 0; nt < 2; ++nt)
            #pragma unroll
            for (int j = 0; j < 2; ++j)
                sacc[nt] = __builtin_amdgcn_wmma_f32_16x16x32_f16(
                    false, qa[j], false, kf[nt][j], (short)0, sacc[nt], false, false);

        #pragma unroll
        for (int r = 0; r < 8; ++r) {
            float v0 = sacc[0][r], v1 = sacc[1][r];
            float mx = fmaxf(v0, v1);
            #pragma unroll
            for (int mask = 8; mask >= 1; mask >>= 1)
                mx = fmaxf(mx, __shfl_xor(mx, mask, 32));
            int rr = rbase + r;
            float m_old = row_m[rr];
            float m_new = fmaxf(m_old, mx);
            float corr  = __expf(m_old - m_new);
            float p0 = __expf(v0 - m_new);
            float p1 = __expf(v1 - m_new);
            float psum = p0 + p1;
            #pragma unroll
            for (int mask = 8; mask >= 1; mask >>= 1)
                psum += __shfl_xor(psum, mask, 32);
            if (col == 0) {
                row_m[rr] = m_new;
                row_l[rr] = row_l[rr] * corr + psum;
            }
            #pragma unroll
            for (int nt = 0; nt < 4; ++nt) oacc[nt][r] *= corr;
            int ln = rr;
            ps[ln * 16 + col]        = (_Float16)p0;
            ps[(ln + 16) * 16 + col] = (_Float16)p1;
        }
        __syncthreads();

        v16h pa = *(const v16h*)&ps[lane * 16];
        #pragma unroll
        for (int nt = 0; nt < 4; ++nt)
            oacc[nt] = __builtin_amdgcn_wmma_f32_16x16x32_f16(
                false, pa, false, vf[nt], (short)0, oacc[nt], false, false);
        __syncthreads();
    }

    #pragma unroll
    for (int nt = 0; nt < 4; ++nt) {
        #pragma unroll
        for (int r = 0; r < 8; ++r) {
            int rr = rbase + r;
            int t  = t0 + rr;
            int d  = nt * 16 + col;
            out[(((size_t)b * T_ + t) * NQ_ + qh) * HD_ + d] =
                (_Float16)(oacc[nt][r] / row_l[rr]);
        }
    }
}

// ---------------------------------------------------------------------------
// LayerNorm, in place on d_out. One 256-thread block per row of 2048.
// ---------------------------------------------------------------------------
__global__ __launch_bounds__(256)
void layernorm_kernel(float* __restrict__ y, const float* __restrict__ gamma,
                      const float* __restrict__ beta)
{
    __shared__ float red[256];
    __shared__ float red2[256];
    int row = blockIdx.x;
    float* p = y + (size_t)row * HT;
    float s = 0.f, s2 = 0.f;
    float vals[8];
    #pragma unroll
    for (int i = 0; i < 8; ++i) {
        float vv = p[threadIdx.x + i * 256];
        vals[i] = vv;
        s += vv; s2 += vv * vv;
    }
    red[threadIdx.x] = s; red2[threadIdx.x] = s2;
    __syncthreads();
    for (int off = 128; off > 0; off >>= 1) {
        if (threadIdx.x < off) {
            red[threadIdx.x]  += red[threadIdx.x + off];
            red2[threadIdx.x] += red2[threadIdx.x + off];
        }
        __syncthreads();
    }
    float mu  = red[0] / (float)HT;
    float var = red2[0] / (float)HT - mu * mu;
    float inv = rsqrtf(var + LN_EPS_);
    #pragma unroll
    for (int i = 0; i < 8; ++i) {
        int c = threadIdx.x + i * 256;
        p[c] = gamma[c] * (vals[i] - mu) * inv + beta[c];
    }
}

// ---------------------------------------------------------------------------
extern "C" void kernel_launch(void* const* d_in, const int* in_sizes, int n_in,
                              void* d_out, int out_size, void* d_ws, size_t ws_size,
                              hipStream_t stream)
{
    const float* tgt = (const float*)d_in[0];
    const float* sme = (const float*)d_in[1];
    const float* mem = (const float*)d_in[2];
    const float* pme = (const float*)d_in[3];
    const float* nme = (const float*)d_in[4];
    const float* Wq  = (const float*)d_in[5];
    const float* bq  = (const float*)d_in[6];
    const float* Wk  = (const float*)d_in[7];
    const float* bk  = (const float*)d_in[8];
    const float* Wv  = (const float*)d_in[9];
    const float* bv  = (const float*)d_in[10];
    const float* Wo  = (const float*)d_in[11];
    const float* bo  = (const float*)d_in[12];
    const float* lng = (const float*)d_in[13];
    const float* lnb = (const float*)d_in[14];
    float* out = (float*)d_out;

    char* ws = (char*)d_ws;
    size_t off = 0;
    _Float16* tgt_h = (_Float16*)(ws + off); off += (size_t)B_ * T_ * HT * 2;  // 16 MB
    _Float16* mem_h = (_Float16*)(ws + off); off += (size_t)B_ * S_ * HT * 2;  // 32 MB
    _Float16* Wq_t  = (_Float16*)(ws + off); off += (size_t)HT * HT * 2;       //  8 MB
    _Float16* Wk_t  = (_Float16*)(ws + off); off += (size_t)HT * DR * 2;       //  2 MB
    _Float16* Wv_t  = (_Float16*)(ws + off); off += (size_t)HT * DR * 2;       //  2 MB
    _Float16* Wo_t  = (_Float16*)(ws + off); off += (size_t)HT * HT * 2;       //  8 MB
    _Float16* q_tmp = (_Float16*)(ws + off); off += (size_t)B_ * T_ * HT * 2;  // 16 MB
    _Float16* q_r   = (_Float16*)(ws + off); off += (size_t)B_ * T_ * HT * 2;  // 16 MB
    _Float16* k_tmp = (_Float16*)(ws + off); off += (size_t)B_ * S_ * DR * 2;  //  8 MB
    _Float16* k_r   = (_Float16*)(ws + off); off += (size_t)B_ * S_ * DR * 2;  //  8 MB
    _Float16* v_t   = (_Float16*)(ws + off); off += (size_t)B_ * S_ * DR * 2;  //  8 MB
    _Float16* ao_h  = (_Float16*)(ws + off); off += (size_t)B_ * T_ * HT * 2;  // 16 MB

    dim3 blk(256);
    // one-time conversions / weight transposes
    cvt_f16_kernel<<<(B_ * T_ * HT) / (8 * 256), blk, 0, stream>>>(tgt, tgt_h, B_ * T_ * HT);
    cvt_f16_kernel<<<(B_ * S_ * HT) / (8 * 256), blk, 0, stream>>>(mem, mem_h, B_ * S_ * HT);
    wtrans_kernel<<<dim3(HT / 64, HT / 32), blk, 0, stream>>>(Wq, Wq_t, HT, HT);
    wtrans_kernel<<<dim3(DR / 64, HT / 32), blk, 0, stream>>>(Wk, Wk_t, HT, DR);
    wtrans_kernel<<<dim3(DR / 64, HT / 32), blk, 0, stream>>>(Wv, Wv_t, HT, DR);
    wtrans_kernel<<<dim3(HT / 64, HT / 32), blk, 0, stream>>>(Wo, Wo_t, HT, HT);
    // Q/K projections (f16, row-major) ; V projection (f16, transposed)
    gemm_wmma_async_kernel<<<dim3(HT / 64, (B_ * T_) / 256), blk, 0, stream>>>(
        tgt_h, Wq_t, bq, q_tmp, nullptr, nullptr, 0.f, nullptr, 0, B_ * T_, HT, HT);
    gemm_wmma_async_kernel<<<dim3(DR / 64, (B_ * S_) / 256), blk, 0, stream>>>(
        mem_h, Wk_t, bk, k_tmp, nullptr, nullptr, 0.f, nullptr, 0, B_ * S_, DR, HT);
    gemm_wmma_async_kernel<<<dim3(DR / 64, (B_ * S_) / 256), blk, 0, stream>>>(
        mem_h, Wv_t, bv, nullptr, nullptr, nullptr, 0.f, v_t, S_, B_ * S_, DR, HT);
    // RoPE (q: identity head map; k: even/odd interleave permutation)
    rope_kernel<<<(B_ * T_ * NQ_ + 255) / 256, blk, 0, stream>>>(
        q_tmp, q_r, sme, sme, B_ * T_, NQ_, 0);
    rope_kernel<<<(B_ * S_ * NKV_ + 255) / 256, blk, 0, stream>>>(
        k_tmp, k_r, pme, nme, B_ * S_, NKV_, 1);
    // grouped-query flash attention (f16 out)
    flash_attn_kernel<<<B_ * NKV_ * GQ * (T_ / 16), dim3(32), 0, stream>>>(
        q_r, k_r, v_t, ao_h);
    // output projection + residual (f32 into d_out)
    gemm_wmma_async_kernel<<<dim3(HT / 64, (B_ * T_) / 256), blk, 0, stream>>>(
        ao_h, Wo_t, bo, nullptr, out, tgt, ALPHA_, nullptr, 0, B_ * T_, HT, HT);
    // layernorm in place
    layernorm_kernel<<<B_ * T_, blk, 0, stream>>>(out, lng, lnb);
}
